// CRF_20126216749360
// MI455X (gfx1250) — matrix operand order repaired and checked by reading, not compiled, not run on hardware
//
#include <hip/hip_runtime.h>
#include <hip/hip_bf16.h>

typedef __attribute__((ext_vector_type(16))) _Float16 v16h;
typedef __attribute__((ext_vector_type(2)))  _Float16 h2;
typedef __attribute__((ext_vector_type(8)))  float    v8f;
typedef __attribute__((ext_vector_type(8)))  int      v8i_t;

#define CRF_B 256
#define CRF_S 2048
#define CRF_T 64
#define LN2F  0.69314718055994531f

static __device__ __forceinline__ int pack2(float a, float b) {
    h2 h;
    h.x = (_Float16)a;
    h.y = (_Float16)b;
    return __builtin_bit_cast(int, h);
}

// ---------------------------------------------------------------------------
// Forward (denominator) kernel, transposed exp-space formulation with a
// table-order K index (equal to the prev-tag index, baked into the
// constant E^T A-fragments) so the C/D output layout IS the next step's
// B-fragment layout: no shuffles, no LDS in the loop.
// Renormalization is deferred one step (power-of-two exponent of the incoming
// state, computed off the critical chain; exact integer bookkeeping in offi).
// ---------------------------------------------------------------------------
__global__ __launch_bounds__(32)
void crf_forward_kernel(const float* __restrict__ em,       // (B,S,T)
                        const float* __restrict__ start_t,  // (T)
                        const float* __restrict__ end_t,    // (T)
                        const float* __restrict__ trans,    // (T,T)
                        const int*   __restrict__ mask,     // (B,S)
                        float* __restrict__ den_out)        // (B)
{
    const int lane = threadIdx.x;           // 0..31
    const int half = lane >> 4;             // 0/1
    const int ln   = lane & 15;
    const int b    = blockIdx.x * 16 + ln;  // this lane's batch column
    const float* __restrict__ emb = em + (size_t)b * CRF_S * CRF_T;

    // ---- constant A fragments: A[j, K] = E^T[j, pi(K)] = exp(trans[pi(K), j])
    // A-layout: element p -> K = 32c + (p&7) + 16*(p>>3) + 8*half.
    // pi is identity (table order):  pi(K) = 32c + 8*half + 16*(p>>3) + (p&7).
    v16h Af[4][2];
#pragma unroll
    for (int u = 0; u < 4; ++u)
#pragma unroll
        for (int c = 0; c < 2; ++c)
#pragma unroll
            for (int p = 0; p < 16; ++p) {
                const int i = 32 * c + 8 * half + 16 * (p >> 3) + (p & 7);
                const int j = 16 * u + ln;
                Af[u][c][p] = (_Float16)__expf(trans[i * CRF_T + j]);
            }

    // ---- X[u][r]: linear alpha for tag j = 16u + 8*half + r, batch b ------
    float X[4][8];
    float offf;                 // initial log offset
    int   offi = 0;             // accumulated power-of-two exponent
    {
        float a0[4][8];
        float mx0 = -3.0e38f;
#pragma unroll
        for (int u = 0; u < 4; ++u)
#pragma unroll
            for (int r = 0; r < 8; ++r) {
                const int j = 16 * u + 8 * half + r;
                a0[u][r] = start_t[j] + emb[j];
                mx0 = fmaxf(mx0, a0[u][r]);
            }
        mx0 = fmaxf(mx0, __shfl_xor(mx0, 16, 32));
#pragma unroll
        for (int u = 0; u < 4; ++u)
#pragma unroll
            for (int r = 0; r < 8; ++r)
                X[u][r] = __expf(a0[u][r] - mx0);
        offf = mx0;
    }

    // ---- prologue: exp(emissions) for step 1 ------------------------------
    float eexp[4][8];
#pragma unroll
    for (int u = 0; u < 4; ++u) {
        const float4 q0 = *(const float4*)(emb + (size_t)CRF_T + 16 * u + 8 * half);
        const float4 q1 = *(const float4*)(emb + (size_t)CRF_T + 16 * u + 8 * half + 4);
        eexp[u][0] = __expf(q0.x); eexp[u][1] = __expf(q0.y);
        eexp[u][2] = __expf(q0.z); eexp[u][3] = __expf(q0.w);
        eexp[u][4] = __expf(q1.x); eexp[u][5] = __expf(q1.y);
        eexp[u][6] = __expf(q1.z); eexp[u][7] = __expf(q1.w);
    }

    for (int s = 1; s < CRF_S; ++s) {
        // issue next step's emission loads early (independent of the chain)
        float nraw[4][8];
        if (s + 1 < CRF_S) {
#pragma unroll
            for (int u = 0; u < 4; ++u) {
                const float* base = emb + (size_t)(s + 1) * CRF_T + 16 * u + 8 * half;
                const float4 q0 = *(const float4*)(base);
                const float4 q1 = *(const float4*)(base + 4);
                nraw[u][0] = q0.x; nraw[u][1] = q0.y; nraw[u][2] = q0.z; nraw[u][3] = q0.w;
                nraw[u][4] = q1.x; nraw[u][5] = q1.y; nraw[u][6] = q1.z; nraw[u][7] = q1.w;
            }
        }
        const int mk = mask[(size_t)b * CRF_S + s];

        // ---- B fragments straight from local registers (thanks to pi) -----
        // element p of chunk c = X[2c + (p>>3)][p&7]
        v16h Bf[2];
#pragma unroll
        for (int c = 0; c < 2; ++c) {
            v8i_t w;
#pragma unroll
            for (int k = 0; k < 4; ++k) {
                w[k]     = pack2(X[2 * c][2 * k],     X[2 * c][2 * k + 1]);
                w[k + 4] = pack2(X[2 * c + 1][2 * k], X[2 * c + 1][2 * k + 1]);
            }
            Bf[c] = __builtin_bit_cast(v16h, w);
        }

        // ---- deferred scale: exponent of incoming state's max (off-chain) --
        float mx = X[0][0];
#pragma unroll
        for (int u = 0; u < 4; ++u)
#pragma unroll
            for (int r = 0; r < 8; ++r)
                mx = fmaxf(mx, X[u][r]);
        mx = fmaxf(mx, __shfl_xor(mx, 16, 32));
        const int   ebits = (__float_as_int(mx) >> 23) & 0xff;    // biased exp
        const float sc    = __int_as_float((251 - ebits) << 23);  // 2^-(e+3)
        const int   einc  = ebits - 124;                          // e + 3
        const bool  upd   = (mk > 0);
#pragma unroll
        for (int u = 0; u < 4; ++u)
#pragma unroll
            for (int r = 0; r < 8; ++r)
                eexp[u][r] *= sc;                                 // esc, off-chain

        // ---- 8 chained WMMAs ---------------------------------------------
        v8f acc[4];
#pragma unroll
        for (int u = 0; u < 4; ++u) {
            v8f z = {};
            z = __builtin_amdgcn_wmma_f32_16x16x32_f16(
                    false, Af[u][0], false, Bf[0], (short)0, z, false, false);
            z = __builtin_amdgcn_wmma_f32_16x16x32_f16(
                    false, Af[u][1], false, Bf[1], (short)0, z, false, false);
            acc[u] = z;
        }

        // ---- masked state update (branchless) -----------------------------
#pragma unroll
        for (int u = 0; u < 4; ++u)
#pragma unroll
            for (int r = 0; r < 8; ++r) {
                const float cand = acc[u][r] * eexp[u][r];
                X[u][r] = upd ? cand : X[u][r];
            }
        offi = upd ? offi + einc : offi;

        // exp() for next step — off the dependence chain
        if (s + 1 < CRF_S) {
#pragma unroll
            for (int u = 0; u < 4; ++u)
#pragma unroll
                for (int r = 0; r < 8; ++r)
                    eexp[u][r] = __expf(nraw[u][r]);
        }
    }

    // ---- denominator: log( sum_j X_j * exp(end_j) ) + offsets -------------
    float ssum = 0.0f;
#pragma unroll
    for (int u = 0; u < 4; ++u)
#pragma unroll
        for (int r = 0; r < 8; ++r)
            ssum += X[u][r] * __expf(end_t[16 * u + 8 * half + r]);
    ssum += __shfl_xor(ssum, 16, 32);
    if (half == 0)
        den_out[b] = __logf(ssum) + offf + (float)offi * LN2F;
}

// ---------------------------------------------------------------------------
// Numerator: one block per batch, threads stride over S, block reduction.
// ---------------------------------------------------------------------------
__global__ __launch_bounds__(256)
void crf_numer_kernel(const float* __restrict__ em,
                      const float* __restrict__ start_t,
                      const float* __restrict__ end_t,
                      const float* __restrict__ trans,
                      const int*   __restrict__ tags,
                      const int*   __restrict__ mask,
                      float* __restrict__ num_out)
{
    const int b = blockIdx.x;
    const int t = threadIdx.x;
    const float* __restrict__ emb = em   + (size_t)b * CRF_S * CRF_T;
    const int*   __restrict__ tgb = tags + (size_t)b * CRF_S;
    const int*   __restrict__ mkb = mask + (size_t)b * CRF_S;

    float part = 0.0f;
    int   cnt  = 0;
    for (int s = t; s < CRF_S; s += 256) {
        const int tg = tgb[s];
        const int mk = mkb[s];
        cnt += (mk != 0) ? 1 : 0;
        if (s > 0) {
            const int tp = tgb[s - 1];
            const float inc = trans[tp * CRF_T + tg] + emb[(size_t)s * CRF_T + tg];
            part += (mk != 0) ? inc : 0.0f;
        }
    }

    __shared__ float sf[256];
    __shared__ int   si[256];
    sf[t] = part;
    si[t] = cnt;
    __syncthreads();
#pragma unroll
    for (int w = 128; w > 0; w >>= 1) {
        if (t < w) { sf[t] += sf[t + w]; si[t] += si[t + w]; }
        __syncthreads();
    }
    if (t == 0) {
        const int tg0 = tgb[0];
        float score = sf[0] + start_t[tg0] + emb[tg0];
        score += end_t[tgb[si[0] - 1]];
        num_out[b] = score;
    }
}

// ---------------------------------------------------------------------------
// Final reduction: out = mean(den - num)
// ---------------------------------------------------------------------------
__global__ __launch_bounds__(256)
void crf_final_kernel(const float* __restrict__ den,
                      const float* __restrict__ num,
                      float* __restrict__ out)
{
    __shared__ float sm[256];
    const int t = threadIdx.x;
    sm[t] = den[t] - num[t];
    __syncthreads();
#pragma unroll
    for (int w = 128; w > 0; w >>= 1) {
        if (t < w) sm[t] += sm[t + w];
        __syncthreads();
    }
    if (t == 0) out[0] = sm[0] / (float)CRF_B;
}

extern "C" void kernel_launch(void* const* d_in, const int* in_sizes, int n_in,
                              void* d_out, int out_size, void* d_ws, size_t ws_size,
                              hipStream_t stream) {
    const float* em      = (const float*)d_in[0];  // (B,S,T)
    const float* start_t = (const float*)d_in[1];  // (T)
    const float* end_t   = (const float*)d_in[2];  // (T)
    const float* trans   = (const float*)d_in[3];  // (T,T)
    const int*   tags    = (const int*)  d_in[4];  // (B,S)
    const int*   mask    = (const int*)  d_in[5];  // (B,S)

    float* den = (float*)d_ws;      // B floats
    float* num = den + CRF_B;       // B floats
    float* out = (float*)d_out;

    crf_forward_kernel<<<CRF_B / 16, 32, 0, stream>>>(em, start_t, end_t, trans,
                                                      mask, den);
    crf_numer_kernel<<<CRF_B, 256, 0, stream>>>(em, start_t, end_t, trans, tags,
                                                mask, num);
    crf_final_kernel<<<1, 256, 0, stream>>>(den, num, out);
}
